// CausalSelfAttention_4690104287621
// MI455X (gfx1250) — compile-verified, matrix operand order checked by reading, not run on hardware
//
#include <hip/hip_runtime.h>
#include <hip/hip_bf16.h>

// ---------------------------------------------------------------------------
// CausalSelfAttention for MI455X (gfx1250), bf16 WMMA, register-blocked.
// B=2, T=2048, C=1024, H=16, D=64
// ---------------------------------------------------------------------------

#define BB   2
#define TT   2048
#define CC   1024
#define HH   16
#define DD   64

typedef __attribute__((ext_vector_type(16))) __bf16 v16bf;
typedef __attribute__((ext_vector_type(8)))  float  v8f;

union Frag16 {
    v16bf v;
    uint4 q[2];
};

__device__ __forceinline__ unsigned short f2bf(float f) {
    unsigned int u = __float_as_uint(f);
    unsigned int r = u + 0x7FFFu + ((u >> 16) & 1u);   // round-to-nearest-even
    return (unsigned short)(r >> 16);
}

__device__ __forceinline__ v8f wmma_bf16(v16bf a, v16bf b, v8f c) {
    return __builtin_amdgcn_wmma_f32_16x16x32_bf16(
        /*neg_a=*/false, a, /*neg_b=*/false, b,
        /*c_mod=*/(short)0, c, /*reuse_a=*/false, /*reuse_b=*/false);
}

// A-layout fragment loader for a 16(row) x 32(K) bf16 tile stored row-major.
// lane L (L<16): row L,    K chunks [0..7]  and [16..23]
// lane L (>=16): row L-16, K chunks [8..15] and [24..31]
__device__ __forceinline__ v16bf load_frag_rm(const unsigned short* base, int ldr,
                                              int row0, int k0, int lane) {
    int r  = row0 + (lane & 15);
    int ko = k0 + ((lane >> 4) << 3);
    Frag16 f;
    const unsigned short* p = base + (size_t)r * ldr + ko;
    f.q[0] = *(const uint4*)(p);
    f.q[1] = *(const uint4*)(p + 16);
    return f.v;
}

__device__ __forceinline__ float red_max16(float v) {
    v = fmaxf(v, __shfl_xor(v, 1));
    v = fmaxf(v, __shfl_xor(v, 2));
    v = fmaxf(v, __shfl_xor(v, 4));
    v = fmaxf(v, __shfl_xor(v, 8));
    return v;
}
__device__ __forceinline__ float red_sum16(float v) {
    v += __shfl_xor(v, 1);
    v += __shfl_xor(v, 2);
    v += __shfl_xor(v, 4);
    v += __shfl_xor(v, 8);
    return v;
}

// ---------------------------------------------------------------------------
// 1) fp32 -> bf16 convert
// ---------------------------------------------------------------------------
__global__ void csa_cvt_bf16(const float* __restrict__ in,
                             unsigned short* __restrict__ out, int n) {
    int i = blockIdx.x * blockDim.x + threadIdx.x;
    if (i < n) out[i] = f2bf(in[i]);
}

// ---------------------------------------------------------------------------
// 2) fp32 [K,N] -> bf16 [N,K] transpose (so GEMM B-operands load row-major)
// ---------------------------------------------------------------------------
__global__ void csa_transpose_bf16(const float* __restrict__ in,
                                   unsigned short* __restrict__ out,
                                   int K, int N) {
    int i = blockIdx.x * blockDim.x + threadIdx.x;   // i = n*K + k
    if (i < K * N) {
        int n = i / K;
        int k = i - n * K;
        out[i] = f2bf(in[(size_t)k * N + n]);
    }
}

// ---------------------------------------------------------------------------
// 3) QKV projection: qkv = Xbf @ Wqkv + b.
//    Per-wave 64x64 output tile: 4 A-frags x 4 B-frags -> 16 WMMAs / k-step.
//    Epilogue scatters Q/K -> [B,H,T,D] bf16, V -> transposed [B,H,D,T] bf16.
// ---------------------------------------------------------------------------
__global__ __launch_bounds__(256) void csa_gemm_qkv(
        const unsigned short* __restrict__ X,    // [B*T, C] bf16
        const unsigned short* __restrict__ Wt,   // [3C, C] bf16 (transposed)
        const float* __restrict__ bias,          // [3C]
        unsigned short* __restrict__ Qb,         // [B,H,T,D]
        unsigned short* __restrict__ Kb,         // [B,H,T,D]
        unsigned short* __restrict__ Vt) {       // [B,H,D,T]
    const int NT = (3 * CC) / 64;                // 48 tiles in N
    int lane = threadIdx.x & 31;
    int wave = threadIdx.x >> 5;
    int tile = blockIdx.x * 8 + wave;            // 64*48 = 3072 tiles
    int mt = tile / NT, nt = tile - mt * NT;
    int m0 = mt * 64, n0 = nt * 64;

    v8f acc[4][4] = {};                          // [mi][ni]
#pragma unroll 4
    for (int k0 = 0; k0 < CC; k0 += 32) {
        v16bf a[4], b[4];
#pragma unroll
        for (int i = 0; i < 4; ++i)
            a[i] = load_frag_rm(X, CC, m0 + 16 * i, k0, lane);
#pragma unroll
        for (int i = 0; i < 4; ++i)
            b[i] = load_frag_rm(Wt, CC, n0 + 16 * i, k0, lane);
        if (k0 + 32 < CC) {                      // cover L2->L0 latency
            __builtin_prefetch(X  + (size_t)(m0 + lane) * CC + k0 + 32, 0, 3);
            __builtin_prefetch(Wt + (size_t)(n0 + lane) * CC + k0 + 32, 0, 3);
        }
#pragma unroll
        for (int mi = 0; mi < 4; ++mi)
#pragma unroll
            for (int ni = 0; ni < 4; ++ni)
                acc[mi][ni] = wmma_bf16(a[mi], b[ni], acc[mi][ni]);
    }

    // epilogue: seg/head are uniform per wave (64-wide tile, 64-aligned)
    int lanecol = lane & 15;
    int mbase   = (lane >> 4) << 3;
    int seg = n0 >> 10;                          // 0=Q 1=K 2=V
    int h   = (n0 & (CC - 1)) >> 6;
#pragma unroll
    for (int ni = 0; ni < 4; ++ni) {
        float bv = bias[n0 + ni * 16 + lanecol];
        int d = ni * 16 + lanecol;
#pragma unroll
        for (int mi = 0; mi < 4; ++mi) {
#pragma unroll
            for (int r = 0; r < 8; ++r) {
                int tglob = m0 + mi * 16 + r + mbase;   // token in [0, B*T)
                int b_    = tglob >> 11;                 // / T
                int t_    = tglob & (TT - 1);
                int bh    = b_ * HH + h;
                unsigned short val = f2bf(acc[mi][ni][r] + bv);
                if (seg == 0)      Qb[((size_t)bh * TT + t_) * DD + d] = val;
                else if (seg == 1) Kb[((size_t)bh * TT + t_) * DD + d] = val;
                else               Vt[((size_t)bh * DD + d) * TT + t_] = val;
            }
        }
    }
}

// ---------------------------------------------------------------------------
// 4) Flash-style causal attention, 32 query rows per wave.
//    K/V fragments for each 32-key tile are shared by two 16-row sub-tiles:
//    16 WMMAs per 16 b128 loads. Online softmax with shfl row reductions;
//    P reaches A-layout via per-wave LDS (in-order DS pipe, wave_barrier).
// ---------------------------------------------------------------------------
__global__ __launch_bounds__(256) void csa_attn(
        const unsigned short* __restrict__ Qb,
        const unsigned short* __restrict__ Kb,
        const unsigned short* __restrict__ Vt,
        unsigned short* __restrict__ Yb) {       // [B,T,C] bf16
    __shared__ __align__(16) unsigned short pbuf[8][2][16 * 32];   // 16 KB

    int lane = threadIdx.x & 31;
    int wave = threadIdx.x >> 5;
    int tile = blockIdx.x * 8 + wave;            // B*H*(T/32) = 2048 tiles
    int bh = tile >> 6;                          // / (T/32)
    int qt = tile & 63;
    int q0 = qt * 32;

    const unsigned short* Qp = Qb + (size_t)bh * TT * DD;
    const unsigned short* Kp = Kb + (size_t)bh * TT * DD;
    const unsigned short* Vp = Vt + (size_t)bh * DD * TT;

    v16bf qf[2][2];                              // [sub][k-half]
#pragma unroll
    for (int s = 0; s < 2; ++s) {
        qf[s][0] = load_frag_rm(Qp, DD, q0 + 16 * s, 0,  lane);
        qf[s][1] = load_frag_rm(Qp, DD, q0 + 16 * s, 32, lane);
    }

    v8f o[2][4] = {};
    float mrow[2][8], lrow[2][8];
#pragma unroll
    for (int s = 0; s < 2; ++s)
#pragma unroll
        for (int r = 0; r < 8; ++r) { mrow[s][r] = -__builtin_inff(); lrow[s][r] = 0.f; }

    const float sm = 0.125f * 1.44269504f;       // 1/sqrt(64) * log2(e)
    int n = lane & 15;
    int mbase = (lane >> 4) << 3;
    int ktiles = (q0 + 31) / 32 + 1;             // wave-uniform causal bound

    for (int kt = 0; kt < ktiles; ++kt) {
        int c0 = kt * 32;

        // ---- shared K fragments for this 32-key tile ----
        v16bf kf[4];
        kf[0] = load_frag_rm(Kp, DD, c0,      0,  lane);
        kf[1] = load_frag_rm(Kp, DD, c0,      32, lane);
        kf[2] = load_frag_rm(Kp, DD, c0 + 16, 0,  lane);
        kf[3] = load_frag_rm(Kp, DD, c0 + 16, 32, lane);

#pragma unroll
        for (int sub = 0; sub < 2; ++sub) {
            // ---- S = Q K^T (two 16x16 tiles over 32 keys, K-dim 64) ----
            v8f s0 = {}, s1 = {};
            s0 = wmma_bf16(qf[sub][0], kf[0], s0);
            s0 = wmma_bf16(qf[sub][1], kf[1], s0);
            s1 = wmma_bf16(qf[sub][0], kf[2], s1);
            s1 = wmma_bf16(qf[sub][1], kf[3], s1);

            // ---- causal mask + online softmax ----
#pragma unroll
            for (int r = 0; r < 8; ++r) {
                int qrow = q0 + sub * 16 + r + mbase;
                float x0 = s0[r] * sm;
                float x1 = s1[r] * sm;
                if (c0 + n      > qrow) x0 = -__builtin_inff();
                if (c0 + 16 + n > qrow) x1 = -__builtin_inff();

                float t = red_max16(fmaxf(x0, x1));
                float mnew  = fmaxf(mrow[sub][r], t);
                float scale = exp2f(mrow[sub][r] - mnew);
                mrow[sub][r] = mnew;

                float p0 = exp2f(x0 - mnew);      // exp2(-inf) == 0
                float p1 = exp2f(x1 - mnew);
                lrow[sub][r] = lrow[sub][r] * scale + red_sum16(p0 + p1);

                o[sub][0][r] *= scale; o[sub][1][r] *= scale;
                o[sub][2][r] *= scale; o[sub][3][r] *= scale;

                int m = r + mbase;
                pbuf[wave][sub][m * 32 + n]      = f2bf(p0);
                pbuf[wave][sub][m * 32 + 16 + n] = f2bf(p1);
            }
        }
        __builtin_amdgcn_wave_barrier();          // LDS pipe is in-order per wave

        // ---- shared V^T fragments, then O += P @ V for both sub-tiles ----
        v16bf vf[4];
        vf[0] = load_frag_rm(Vp, TT, 0,  c0, lane);
        vf[1] = load_frag_rm(Vp, TT, 16, c0, lane);
        vf[2] = load_frag_rm(Vp, TT, 32, c0, lane);
        vf[3] = load_frag_rm(Vp, TT, 48, c0, lane);

        int rr = lane & 15;
        int ko = (lane >> 4) << 3;
#pragma unroll
        for (int sub = 0; sub < 2; ++sub) {
            Frag16 pf;
            pf.q[0] = *(const uint4*)&pbuf[wave][sub][rr * 32 + ko];
            pf.q[1] = *(const uint4*)&pbuf[wave][sub][rr * 32 + 16 + ko];
#pragma unroll
            for (int j = 0; j < 4; ++j)
                o[sub][j] = wmma_bf16(pf.v, vf[j], o[sub][j]);
        }
        __builtin_amdgcn_wave_barrier();
    }

    // ---- finalize and store y to [B,T,C] bf16 ----
    int h  = bh & (HH - 1);
    int b_ = bh >> 4;
#pragma unroll
    for (int sub = 0; sub < 2; ++sub) {
#pragma unroll
        for (int r = 0; r < 8; ++r) {
            float inv = 1.0f / lrow[sub][r];
            int t_ = q0 + sub * 16 + r + mbase;
            size_t rowoff = ((size_t)b_ * TT + t_) * CC + h * DD;
            Yb[rowoff +  0 + n] = f2bf(o[sub][0][r] * inv);
            Yb[rowoff + 16 + n] = f2bf(o[sub][1][r] * inv);
            Yb[rowoff + 32 + n] = f2bf(o[sub][2][r] * inv);
            Yb[rowoff + 48 + n] = f2bf(o[sub][3][r] * inv);
        }
    }
}

// ---------------------------------------------------------------------------
// 5) Output projection: out(fp32) = Ybf @ Wproj + b_proj. 64x64 wave tiles.
// ---------------------------------------------------------------------------
__global__ __launch_bounds__(256) void csa_gemm_proj(
        const unsigned short* __restrict__ Y,    // [B*T, C] bf16
        const unsigned short* __restrict__ Wt,   // [C, C] bf16 (transposed)
        const float* __restrict__ bias,          // [C]
        float* __restrict__ out) {               // [B*T, C] fp32
    const int NT = CC / 64;                      // 16
    int lane = threadIdx.x & 31;
    int wave = threadIdx.x >> 5;
    int tile = blockIdx.x * 8 + wave;            // 64*16 = 1024 tiles
    int mt = tile / NT, nt = tile - mt * NT;
    int m0 = mt * 64, n0 = nt * 64;

    v8f acc[4][4] = {};
#pragma unroll 4
    for (int k0 = 0; k0 < CC; k0 += 32) {
        v16bf a[4], b[4];
#pragma unroll
        for (int i = 0; i < 4; ++i)
            a[i] = load_frag_rm(Y, CC, m0 + 16 * i, k0, lane);
#pragma unroll
        for (int i = 0; i < 4; ++i)
            b[i] = load_frag_rm(Wt, CC, n0 + 16 * i, k0, lane);
        if (k0 + 32 < CC) {
            __builtin_prefetch(Y  + (size_t)(m0 + lane) * CC + k0 + 32, 0, 3);
            __builtin_prefetch(Wt + (size_t)(n0 + lane) * CC + k0 + 32, 0, 3);
        }
#pragma unroll
        for (int mi = 0; mi < 4; ++mi)
#pragma unroll
            for (int ni = 0; ni < 4; ++ni)
                acc[mi][ni] = wmma_bf16(a[mi], b[ni], acc[mi][ni]);
    }

    int lanecol = lane & 15;
    int mbase   = (lane >> 4) << 3;
#pragma unroll
    for (int ni = 0; ni < 4; ++ni) {
        int ncol = n0 + ni * 16 + lanecol;
        float bv = bias[ncol];
#pragma unroll
        for (int mi = 0; mi < 4; ++mi) {
#pragma unroll
            for (int r = 0; r < 8; ++r) {
                int m = m0 + mi * 16 + r + mbase;
                out[(size_t)m * CC + ncol] = acc[mi][ni][r] + bv;
            }
        }
    }
}

// ---------------------------------------------------------------------------
// launch
// ---------------------------------------------------------------------------
extern "C" void kernel_launch(void* const* d_in, const int* in_sizes, int n_in,
                              void* d_out, int out_size, void* d_ws, size_t ws_size,
                              hipStream_t stream) {
    const float* x     = (const float*)d_in[0];   // [B,T,C]
    const float* Wqkv  = (const float*)d_in[1];   // [C,3C]
    const float* bqkv  = (const float*)d_in[2];   // [3C]
    const float* Wproj = (const float*)d_in[3];   // [C,C]
    const float* bproj = (const float*)d_in[4];   // [C]
    float* out = (float*)d_out;

    const size_t nX   = (size_t)BB * TT * CC;         // 4,194,304
    const size_t nWq  = (size_t)CC * 3 * CC;          // 3,145,728
    const size_t nWp  = (size_t)CC * CC;              // 1,048,576
    const size_t nHd  = (size_t)BB * HH * TT * DD;    // 4,194,304

    unsigned short* ws = (unsigned short*)d_ws;
    size_t off = 0;
    unsigned short* Xbf    = ws + off; off += nX;
    unsigned short* Wqkvt  = ws + off; off += nWq;
    unsigned short* Wprojt = ws + off; off += nWp;
    unsigned short* Qb     = ws + off; off += nHd;
    unsigned short* Kb     = ws + off; off += nHd;
    unsigned short* Vt     = ws + off; off += nHd;
    unsigned short* Yb     = ws + off; off += nX;     // ~50.3 MB total

    csa_cvt_bf16<<<(int)((nX + 255) / 256), 256, 0, stream>>>(x, Xbf, (int)nX);
    csa_transpose_bf16<<<(int)((nWq + 255) / 256), 256, 0, stream>>>(Wqkv, Wqkvt, CC, 3 * CC);
    csa_transpose_bf16<<<(int)((nWp + 255) / 256), 256, 0, stream>>>(Wproj, Wprojt, CC, CC);

    // (B*T/64)*(3C/64) = 3072 wave tiles / 8 waves = 384 blocks
    csa_gemm_qkv<<<384, 256, 0, stream>>>(Xbf, Wqkvt, bqkv, Qb, Kb, Vt);
    // B*H*(T/32) = 2048 wave tiles / 8 waves = 256 blocks
    csa_attn<<<256, 256, 0, stream>>>(Qb, Kb, Vt, Yb);
    // (B*T/64)*(C/64) = 1024 wave tiles / 8 waves = 128 blocks
    csa_gemm_proj<<<128, 256, 0, stream>>>(Yb, Wprojt, bproj, out);
}
